// BiLSTMCRF_51445118272135
// MI455X (gfx1250) — matrix operand order repaired and checked by reading, not compiled, not run on hardware
//
#include <hip/hip_runtime.h>
#include <hip/hip_bf16.h>

#define DEVI __device__ __forceinline__

typedef __attribute__((ext_vector_type(16))) __bf16 v16bf;
typedef __attribute__((ext_vector_type(8)))  float  v8f;
typedef int v4i_ __attribute__((vector_size(16)));   // matches builtin param type

constexpr int kB = 64, kT = 512, kE = 256, kH = 256, kG = 1024, kL = 9;
constexpr int kM = kB * kT;            // 32768 (B*T rows)
constexpr int kLogitsN = kM * kL;      // 294912

// ---------------------------------------------------------------------------
// Async global->LDS copy (CDNA5 GLOBAL_LOAD_ASYNC_TO_LDS_B128, ASYNCcnt).
// Guarded by __has_builtin so both ROCm 7.2 and amdgpu-toolchain compile.
// Probe-learned signature: (AS1 v4i* gsrc, AS3 v4i* ldst, imm offset, imm cpol)
// ---------------------------------------------------------------------------
#if defined(__has_builtin)
#if __has_builtin(__builtin_amdgcn_global_load_async_to_lds_b128) && \
    __has_builtin(__builtin_amdgcn_s_wait_asynccnt)
#define USE_ASYNC_LDS 1
#endif
#endif

DEVI void async_copy_b128(void* lds_dst, const void* gsrc) {
#ifdef USE_ASYNC_LDS
  typedef __attribute__((address_space(1))) v4i_* gp_t;
  typedef __attribute__((address_space(3))) v4i_* lp_t;
  __builtin_amdgcn_global_load_async_to_lds_b128(
      (gp_t)(uintptr_t)gsrc, (lp_t)(uintptr_t)lds_dst, 0, 0);
#else
  *(uint4*)lds_dst = *(const uint4*)gsrc;
#endif
}

DEVI void async_copy_fence() {
#ifdef USE_ASYNC_LDS
  __builtin_amdgcn_s_wait_asynccnt(0);
#endif
}

// ---------------------------------------------------------------------------
// WMMA fragment helpers (bf16 16x16x32, wave32)
// A: 16x32 row-major source. lane = g*16+m holds row m, K = {g*8..g*8+7, 16+g*8..16+g*8+7}
// B: stored column-major (N x K row-major "WT"). lane = g*16+n holds col n, K = {g*8..g*8+7, 16+g*8..16+g*8+7}
// ---------------------------------------------------------------------------
DEVI v16bf load_frag_a(const __hip_bfloat16* p, int lda, int lane) {
  const int g = lane >> 4, m = lane & 15;
  const __hip_bfloat16* q = p + (size_t)m * (size_t)lda + g * 8;
  uint4 u[2];
  u[0] = *(const uint4*)(q);        // K = g*8 .. g*8+7
  u[1] = *(const uint4*)(q + 16);   // K = 16+g*8 .. 16+g*8+7
  v16bf r; __builtin_memcpy(&r, u, sizeof(r)); return r;
}

DEVI v16bf load_frag_b(const __hip_bfloat16* p, int ldb, int lane) {
  const int g = lane >> 4, n = lane & 15;
  const __hip_bfloat16* q = p + (size_t)n * (size_t)ldb + g * 8;
  uint4 u[2];
  u[0] = *(const uint4*)(q);        // K = g*8 .. g*8+7
  u[1] = *(const uint4*)(q + 16);   // K = 16+g*8 .. 16+g*8+7
  v16bf r; __builtin_memcpy(&r, u, sizeof(r)); return r;
}

DEVI v8f wmma_bf16(v16bf a, v16bf b, v8f c) {
  return __builtin_amdgcn_wmma_f32_16x16x32_bf16(false, a, false, b, (short)0, c,
                                                 false, false);
}

DEVI float sigmoidf(float x) { return 1.0f / (1.0f + __expf(-x)); }

// ---------------------------------------------------------------------------
// Prep kernels
// ---------------------------------------------------------------------------
__global__ void init_counters_kernel(int* c) {
  if (threadIdx.x < 2) c[threadIdx.x] = 0;
}

__global__ void seqlen_kernel(const int* __restrict__ tokens,
                              int* __restrict__ sl_i, float* __restrict__ sl_f) {
  __shared__ int red[256];
  const int b = blockIdx.x, tid = threadIdx.x;
  int cnt = 0;
  for (int t = tid; t < kT; t += 256) cnt += (tokens[b * kT + t] != 0);
  red[tid] = cnt; __syncthreads();
  for (int s = 128; s > 0; s >>= 1) {
    if (tid < s) red[tid] += red[tid + s];
    __syncthreads();
  }
  if (tid == 0) { sl_i[b] = red[0]; sl_f[b] = (float)red[0]; }
}

// emb_bf16[b,t,:] = bf16(emb_table[tokens[b,t],:])
__global__ void embed_kernel(const int* __restrict__ tokens,
                             const float* __restrict__ table,
                             __hip_bfloat16* __restrict__ out) {
  const int r = blockIdx.x;                 // 0..B*T-1
  const int tok = tokens[r];
  out[(size_t)r * kE + threadIdx.x] =
      __float2bfloat16(table[(size_t)tok * kE + threadIdx.x]);
}

// dst[n*K+k] = bf16(src[k*N+n])   (K x N -> N x K, i.e. column-major bf16)
__global__ void transpose_bf16_kernel(const float* __restrict__ src,
                                      __hip_bfloat16* __restrict__ dst,
                                      int K, int N) {
  const int idx = blockIdx.x * 256 + threadIdx.x;
  if (idx >= K * N) return;
  const int n = idx / K, k = idx - n * K;
  dst[idx] = __float2bfloat16(src[(size_t)k * N + n]);
}

// W_dense (512 x 9) -> padded column-major bf16 [16][512]
__global__ void wdense_pad_kernel(const float* __restrict__ Wd,
                                  __hip_bfloat16* __restrict__ dst) {
  const int idx = blockIdx.x * 256 + threadIdx.x;   // 16*512
  const int n = idx >> 9, k = idx & 511;
  const float v = (n < kL) ? Wd[(size_t)k * kL + n] : 0.0f;
  dst[idx] = __float2bfloat16(v);
}

// ---------------------------------------------------------------------------
// GEMM1: xg = emb @ Wx + b     (M=32768, N=1024, K=256) bf16 WMMA, f32 acc
// block = 256 threads (8 waves, 2x4), tile 128x128, wave tile 64x32
// ---------------------------------------------------------------------------
__global__ __launch_bounds__(256) void gemm_xg_kernel(
    const __hip_bfloat16* __restrict__ X,   // [M][256]
    const __hip_bfloat16* __restrict__ WT,  // [1024][256] col-major of Wx
    const float* __restrict__ bias,         // [1024]
    float* __restrict__ Y) {                // [M][1024]
  const int lane = threadIdx.x & 31, w = threadIdx.x >> 5;
  const int wm = w >> 2, wn = w & 3;
  const int m0 = blockIdx.x * 128 + wm * 64;
  const int n0 = blockIdx.y * 128 + wn * 32;

  v8f acc[4][2] = {};
  for (int k0 = 0; k0 < 256; k0 += 32) {
    if (k0 + 32 < 256)
      __builtin_prefetch(X + (size_t)m0 * 256 + k0 + 32, 0, 1);
    v16bf a[4], bb[2];
#pragma unroll
    for (int i = 0; i < 4; ++i)
      a[i] = load_frag_a(X + (size_t)(m0 + i * 16) * 256 + k0, 256, lane);
#pragma unroll
    for (int j = 0; j < 2; ++j)
      bb[j] = load_frag_b(WT + (size_t)(n0 + j * 16) * 256 + k0, 256, lane);
#pragma unroll
    for (int i = 0; i < 4; ++i)
#pragma unroll
      for (int j = 0; j < 2; ++j) acc[i][j] = wmma_bf16(a[i], bb[j], acc[i][j]);
  }
  const int g = lane >> 4, nn = lane & 15;
#pragma unroll
  for (int i = 0; i < 4; ++i)
#pragma unroll
    for (int j = 0; j < 2; ++j) {
      const int nb = n0 + j * 16 + nn;
      const float bv = bias[nb];
#pragma unroll
      for (int v = 0; v < 8; ++v)
        Y[(size_t)(m0 + i * 16 + 8 * g + v) * kG + nb] = acc[i][j][v] + bv;
    }
}

// ---------------------------------------------------------------------------
// Persistent bidirectional LSTM recurrence.
// grid = 8 WGs: blocks 0..3 = forward (t ascending), 4..7 = backward (descending).
// Each WG owns 64 h-columns; its Wh slice (256 K x 256 local gate cols, bf16,
// column-major) is staged into LDS (128KB) via async global->LDS b128 copies.
// gbuf 64x256 f32 (64KB), c-state 64x64 f32. Device-scope spin barrier
// (one counter per direction) syncs the 4 WGs per timestep.
// ---------------------------------------------------------------------------
DEVI void grid_barrier(int* cnt, int target) {
  __threadfence();
  __syncthreads();
  if (threadIdx.x == 0) {
    __hip_atomic_fetch_add(cnt, 1, __ATOMIC_RELEASE, __HIP_MEMORY_SCOPE_AGENT);
    while (__hip_atomic_load(cnt, __ATOMIC_ACQUIRE, __HIP_MEMORY_SCOPE_AGENT) <
           target)
      __builtin_amdgcn_s_sleep(1);
  }
  __syncthreads();
}

constexpr size_t kSmemWh   = (size_t)256 * 256 * 2;   // 128 KB
constexpr size_t kSmemGbuf = (size_t)64 * 256 * 4;    //  64 KB
constexpr size_t kSmemC    = (size_t)64 * 64 * 4;     //  16 KB
constexpr size_t kSmemRecur = kSmemWh + kSmemGbuf + kSmemC;

__global__ __launch_bounds__(256) void lstm_recur_kernel(
    const float* __restrict__ xg_f, const float* __restrict__ xg_b,
    const __hip_bfloat16* __restrict__ WhT_f,
    const __hip_bfloat16* __restrict__ WhT_b,
    __hip_bfloat16* __restrict__ h_f, __hip_bfloat16* __restrict__ h_b,
    int* __restrict__ counters) {
  extern __shared__ char smem[];
  __hip_bfloat16* WhL = (__hip_bfloat16*)smem;            // [256 local cols][256 K]
  float* gbuf = (float*)(smem + kSmemWh);                 // [64 b][256 local col]
  float* cbuf = (float*)(smem + kSmemWh + kSmemGbuf);     // [64 b][64 j]

  const int tid = threadIdx.x;
  const int lane = tid & 31, w = tid >> 5;
  const int wm = w & 3;        // batch tile (16 rows)
  const int half = w >> 2;     // local-col half (128 cols)
  const int dir = blockIdx.x >> 2;
  const int wg  = blockIdx.x & 3;
  const int n0  = wg * 64;     // owned h-columns [n0, n0+64)

  const float* xg = dir ? xg_b : xg_f;
  const __hip_bfloat16* WhT = dir ? WhT_b : WhT_f;
  __hip_bfloat16* hout = dir ? h_b : h_f;
  int* cnt = counters + dir;

  // Stage Wh slice into LDS via async b128 copies, reordered so local col
  // c = q*64+j maps to global gate col q*256+n0+j.
  {
    const uint4* src = (const uint4*)WhT;   // rows of 256 bf16 = 32 uint4
    uint4* dst = (uint4*)WhL;
    for (int it = 0; it < 32; ++it) {
      const int lin = it * 256 + tid;       // 8192 uint4 total
      const int c = lin >> 5, u = lin & 31;
      const int G = ((c >> 6) << 8) + n0 + (c & 63);
      async_copy_b128(dst + (size_t)c * 32 + u, src + (size_t)G * 32 + u);
    }
    async_copy_fence();
  }
  // c state init
  for (int s = 0; s < 16; ++s) cbuf[tid + s * 256] = 0.0f;
  __syncthreads();

  for (int iter = 0; iter < kT; ++iter) {
    const int t = dir ? (kT - 1 - iter) : iter;
    const int t_prev = dir ? (t + 1) : (t - 1);

    v8f acc[8] = {};
    if (iter > 0) {
      const __hip_bfloat16* Abase =
          hout + (size_t)(wm * 16) * (size_t)(kT * kH) + (size_t)t_prev * kH;
#pragma unroll 1
      for (int k0 = 0; k0 < 256; k0 += 32) {
        const v16bf a = load_frag_a(Abase + k0, kT * kH, lane);
#pragma unroll
        for (int nt = 0; nt < 8; ++nt) {
          const v16bf bfr =
              load_frag_b(WhL + (size_t)(half * 128 + nt * 16) * 256 + k0, 256, lane);
          acc[nt] = wmma_bf16(a, bfr, acc[nt]);
        }
      }
    }
    // spill gate pre-activations to LDS
    {
      const int g = lane >> 4, nn = lane & 15;
      const int Mb = wm * 16 + 8 * g;
#pragma unroll
      for (int nt = 0; nt < 8; ++nt) {
        const int Cc = half * 128 + nt * 16 + nn;
#pragma unroll
        for (int v = 0; v < 8; ++v) gbuf[(size_t)(Mb + v) * 256 + Cc] = acc[nt][v];
      }
    }
    __syncthreads();

    // cell update for owned 64x64 (batch x j)
    for (int s = 0; s < 16; ++s) {
      const int e = tid + s * 256;
      const int b = e >> 6, j = e & 63;
      const size_t xbase = ((size_t)b * kT + t) * kG + n0 + j;
      const float gi = gbuf[b * 256 + 0 * 64 + j] + xg[xbase + 0 * 256];
      const float gf = gbuf[b * 256 + 1 * 64 + j] + xg[xbase + 1 * 256];
      const float gg = gbuf[b * 256 + 2 * 64 + j] + xg[xbase + 2 * 256];
      const float go = gbuf[b * 256 + 3 * 64 + j] + xg[xbase + 3 * 256];
      const float cn = sigmoidf(gf) * cbuf[e] + sigmoidf(gi) * tanhf(gg);
      cbuf[e] = cn;
      const float hh = sigmoidf(go) * tanhf(cn);
      hout[((size_t)b * kT + t) * kH + n0 + j] = __float2bfloat16(hh);
    }

    grid_barrier(cnt, (iter + 1) * 4);
  }
}

// ---------------------------------------------------------------------------
// logits = concat(h_f, h_b) @ W_dense + b_dense   (M=32768, K=512, N=9 pad 16)
// ---------------------------------------------------------------------------
__global__ __launch_bounds__(256) void gemm_logits_kernel(
    const __hip_bfloat16* __restrict__ hF, const __hip_bfloat16* __restrict__ hB,
    const __hip_bfloat16* __restrict__ WdT,  // [16][512] padded col-major
    const float* __restrict__ bd, float* __restrict__ out) {
  const int lane = threadIdx.x & 31, w = threadIdx.x >> 5;
  const int m0 = blockIdx.x * 128 + w * 16;
  v8f acc = {};
  for (int k0 = 0; k0 < 512; k0 += 32) {
    const __hip_bfloat16* src = (k0 < 256) ? hF : hB;
    const int kk = k0 & 255;
    const v16bf a = load_frag_a(src + (size_t)m0 * 256 + kk, 256, lane);
    const v16bf bfr = load_frag_b(WdT + k0, 512, lane);
    acc = wmma_bf16(a, bfr, acc);
  }
  const int g = lane >> 4, nn = lane & 15;
  if (nn < kL) {
    const float bv = bd[nn];
#pragma unroll
    for (int v = 0; v < 8; ++v)
      out[(size_t)(m0 + 8 * g + v) * kL + nn] = acc[v] + bv;
  }
}

// ---------------------------------------------------------------------------
// CRF log-likelihood: one wave32 per batch row; lane j = state j (j<9)
// ---------------------------------------------------------------------------
__global__ __launch_bounds__(32) void crf_kernel(
    const float* __restrict__ logits, const int* __restrict__ labels,
    const int* __restrict__ sl, const float* __restrict__ trans,
    float* __restrict__ ll_out) {
  const int b = blockIdx.x, j = threadIdx.x;
  const int len = sl[b];
  const float* lg = logits + (size_t)b * kT * kL;

  float tj[kL];
#pragma unroll
  for (int i = 0; i < kL; ++i) tj[i] = (j < kL) ? trans[i * kL + j] : 0.0f;

  float alpha = (j < kL) ? lg[j] : -1e30f;
  for (int t = 1; t < kT; ++t) {
    float ai[kL];
#pragma unroll
    for (int i = 0; i < kL; ++i) ai[i] = __shfl(alpha, i, 32);
    float mx = -1e30f;
#pragma unroll
    for (int i = 0; i < kL; ++i) mx = fmaxf(mx, ai[i] + tj[i]);
    float s = 0.0f;
#pragma unroll
    for (int i = 0; i < kL; ++i) s += __expf(ai[i] + tj[i] - mx);
    const float nv = mx + __logf(s) + ((j < kL) ? lg[t * kL + j] : 0.0f);
    if (t < len && j < kL) alpha = nv;
  }
  // log_norm over states
  float aj[kL];
#pragma unroll
  for (int i = 0; i < kL; ++i) aj[i] = __shfl(alpha, i, 32);
  float mx = -1e30f;
#pragma unroll
  for (int i = 0; i < kL; ++i) mx = fmaxf(mx, aj[i]);
  float s = 0.0f;
#pragma unroll
  for (int i = 0; i < kL; ++i) s += __expf(aj[i] - mx);
  const float log_norm = mx + __logf(s);

  // gold score (lanes stride over t)
  const int* tg = labels + (size_t)b * kT;
  float us = 0.0f, bs = 0.0f;
  for (int t = j; t < kT; t += 32) {
    if (t < len) {
      us += lg[t * kL + tg[t]];
      if (t >= 1) bs += trans[tg[t - 1] * kL + tg[t]];
    }
  }
  for (int off = 16; off > 0; off >>= 1) {
    us += __shfl_down(us, off, 32);
    bs += __shfl_down(bs, off, 32);
  }
  if (j == 0) ll_out[b] = (us + bs) - log_norm;
}

// ---------------------------------------------------------------------------
// Host launch
// ---------------------------------------------------------------------------
static constexpr size_t alignup(size_t x) { return (x + 255) & ~(size_t)255; }

extern "C" void kernel_launch(void* const* d_in, const int* in_sizes, int n_in,
                              void* d_out, int out_size, void* d_ws, size_t ws_size,
                              hipStream_t stream) {
  (void)in_sizes; (void)n_in; (void)out_size; (void)ws_size;
  const int*   tokens = (const int*)d_in[0];
  const int*   labels = (const int*)d_in[1];
  const float* table  = (const float*)d_in[2];
  const float* Wx_f   = (const float*)d_in[3];
  const float* Wh_f   = (const float*)d_in[4];
  const float* b_f    = (const float*)d_in[5];
  const float* Wx_b   = (const float*)d_in[6];
  const float* Wh_b   = (const float*)d_in[7];
  const float* b_b    = (const float*)d_in[8];
  const float* Wd     = (const float*)d_in[9];
  const float* bd     = (const float*)d_in[10];
  const float* trans  = (const float*)d_in[11];
  float* out = (float*)d_out;               // [logits | seq_len | ll]
  char* ws = (char*)d_ws;

  size_t off = 0;
  __hip_bfloat16* embB = (__hip_bfloat16*)(ws + off); off = alignup(off + (size_t)kM * kE * 2);
  __hip_bfloat16* WxTf = (__hip_bfloat16*)(ws + off); off = alignup(off + (size_t)kG * kE * 2);
  __hip_bfloat16* WhTf = (__hip_bfloat16*)(ws + off); off = alignup(off + (size_t)kG * kH * 2);
  __hip_bfloat16* WxTb = (__hip_bfloat16*)(ws + off); off = alignup(off + (size_t)kG * kE * 2);
  __hip_bfloat16* WhTb = (__hip_bfloat16*)(ws + off); off = alignup(off + (size_t)kG * kH * 2);
  __hip_bfloat16* WdT  = (__hip_bfloat16*)(ws + off); off = alignup(off + (size_t)16 * 512 * 2);
  float* xg_f = (float*)(ws + off); off = alignup(off + (size_t)kM * kG * 4);
  float* xg_b = (float*)(ws + off); off = alignup(off + (size_t)kM * kG * 4);
  __hip_bfloat16* h_f = (__hip_bfloat16*)(ws + off); off = alignup(off + (size_t)kM * kH * 2);
  __hip_bfloat16* h_b = (__hip_bfloat16*)(ws + off); off = alignup(off + (size_t)kM * kH * 2);
  int* sl_i = (int*)(ws + off); off = alignup(off + kB * 4);
  int* counters = (int*)(ws + off); off = alignup(off + 2 * 4);

  float* seqlen_out = out + kLogitsN;
  float* ll_out     = out + kLogitsN + kB;

  init_counters_kernel<<<1, 32, 0, stream>>>(counters);
  seqlen_kernel<<<kB, 256, 0, stream>>>(tokens, sl_i, seqlen_out);
  embed_kernel<<<kM, kE, 0, stream>>>(tokens, table, embB);

  const int trN = (kE * kG + 255) / 256;
  transpose_bf16_kernel<<<trN, 256, 0, stream>>>(Wx_f, WxTf, kE, kG);
  transpose_bf16_kernel<<<trN, 256, 0, stream>>>(Wh_f, WhTf, kH, kG);
  transpose_bf16_kernel<<<trN, 256, 0, stream>>>(Wx_b, WxTb, kE, kG);
  transpose_bf16_kernel<<<trN, 256, 0, stream>>>(Wh_b, WhTb, kH, kG);
  wdense_pad_kernel<<<(16 * 512) / 256, 256, 0, stream>>>(Wd, WdT);

  dim3 g1(kM / 128, kG / 128);
  gemm_xg_kernel<<<g1, 256, 0, stream>>>(embB, WxTf, b_f, xg_f);
  gemm_xg_kernel<<<g1, 256, 0, stream>>>(embB, WxTb, b_b, xg_b);

  lstm_recur_kernel<<<8, 256, kSmemRecur, stream>>>(xg_f, xg_b, WhTf, WhTb,
                                                    h_f, h_b, counters);

  gemm_logits_kernel<<<kM / 128, 256, 0, stream>>>(h_f, h_b, WdT, bd, out);
  crf_kernel<<<kB, 32, 0, stream>>>(out, labels, sl_i, trans, ll_out);
}
